// ZoomAttention_91216515432970
// MI455X (gfx1250) — compile-verified, matrix operand order checked by reading, not run on hardware
//
#include <hip/hip_runtime.h>
#include <hip/hip_bf16.h>
#include <stdint.h>
#include <math.h>

// ---------------------------------------------------------------------------
// ZoomAttention forward for MI455X (gfx1250, wave32, WMMA bf16 path)
//   - bf16 WMMA for all matrix math (f32 accumulate)
//   - flash-style attention (scores never hit HBM)
//   - async-to-LDS double buffering of K/V tiles (ASYNCcnt pipeline)
//   - ping-pong (unroll-by-2) register double buffering in GEMMs: no mov rotation
// ---------------------------------------------------------------------------

typedef __attribute__((ext_vector_type(16))) __bf16 bf16x16;
typedef __attribute__((ext_vector_type(8)))  float  f32x8;
typedef __attribute__((ext_vector_type(4)))  int    i32x4;

#define B_DIM   2
#define N_SEQ   2048
#define D_MODEL 1024
#define H_NUM   16
#define HD      64

// ---------------------------------------------------------------------------
// gfx1250 async load-to-LDS (ASYNCcnt path), with safe fallback
// ---------------------------------------------------------------------------
#if defined(__has_builtin)
#if __has_builtin(__builtin_amdgcn_global_load_async_to_lds_b128)
#define HAS_ASYNC 1
#endif
#endif

__device__ __forceinline__ void async_copy16(const void* g, void* l) {
#ifdef HAS_ASYNC
    typedef __attribute__((address_space(1))) i32x4 gvec_t;
    typedef __attribute__((address_space(3))) i32x4 lvec_t;
    __builtin_amdgcn_global_load_async_to_lds_b128(
        (gvec_t*)(g), (lvec_t*)(l), 0, 0);
#else
    *(uint4*)l = *(const uint4*)g;
#endif
}

#ifdef HAS_ASYNC
// outstanding <= 16: the *previous* 16-instruction tile copy has completed
#define WAIT_ASYNC_PREV() asm volatile("s_wait_asynccnt 16" ::: "memory")
#define WAIT_ASYNC_ALL()  asm volatile("s_wait_asynccnt 0"  ::: "memory")
#else
#define WAIT_ASYNC_PREV() __syncthreads()
#define WAIT_ASYNC_ALL()  __syncthreads()
#endif

// ---------------------------------------------------------------------------
// WMMA helpers
// ---------------------------------------------------------------------------

// 16x32 bf16 A-operand fragment load (also valid for the dense 32x16 B-operand,
// whose per-lane layout mirrors A with column<->row), from a row-major matrix
// where the K dimension is contiguous. ld = row stride in elements (even).
// Per ISA 7.12.2: lane L holds row (L&15); dword j holds K-pair
//   Kpair = ((j<4) ? j : 4+j) + 4*(L>=16).
__device__ __forceinline__ bf16x16 load_frag(const __bf16* base, int ld) {
    const int lane = threadIdx.x & 31;
    const int row  = lane & 15;
    const int kh   = lane >> 4;
    const uint32_t* p = (const uint32_t*)(base + (size_t)row * (size_t)ld);
    union { bf16x16 v; uint32_t u[8]; } r;
#pragma unroll
    for (int j = 0; j < 8; ++j) {
        const int kp = ((j < 4) ? j : (4 + j)) + 4 * kh;
        r.u[j] = p[kp];
    }
    return r.v;
}

__device__ __forceinline__ f32x8 wmma_bf16(bf16x16 a, bf16x16 b, f32x8 c) {
    // (neg_a, A, neg_b, B, c_mod, C, reuse_a, reuse_b)
    return __builtin_amdgcn_wmma_f32_16x16x32_bf16(false, a, false, b,
                                                   (short)0, c, false, false);
}

__device__ __forceinline__ f32x8 zero8() {
    f32x8 z = {0.f, 0.f, 0.f, 0.f, 0.f, 0.f, 0.f, 0.f};
    return z;
}

// Reductions across the 16-lane half-groups {0..15} and {16..31}
// (xor masks 1,2,4,8 never cross bit 4 -> halves stay separate).
__device__ __forceinline__ float hmax16(float v) {
#pragma unroll
    for (int m = 1; m <= 8; m <<= 1) v = fmaxf(v, __shfl_xor(v, m, 32));
    return v;
}
__device__ __forceinline__ float hsum16(float v) {
#pragma unroll
    for (int m = 1; m <= 8; m <<= 1) v += __shfl_xor(v, m, 32);
    return v;
}

// ---------------------------------------------------------------------------
// Precision conversion kernels
// ---------------------------------------------------------------------------

__global__ void k_cvt_bf16(const float* __restrict__ in, __bf16* __restrict__ out, int n) {
    int i = blockIdx.x * 256 + threadIdx.x;
    if (i < n) out[i] = (__bf16)in[i];
}

// w: [Krows, Ccols] f32 row-major  ->  wt: [Ccols, Krows] bf16 row-major
__global__ void k_transpose_cvt(const float* __restrict__ w, __bf16* __restrict__ wt,
                                int Krows, int Ccols) {
    int idx = blockIdx.x * 256 + threadIdx.x;
    if (idx < Krows * Ccols) {
        int c = idx / Krows;
        int k = idx - c * Krows;
        wt[idx] = (__bf16)w[(size_t)k * Ccols + c];
    }
}

// ---------------------------------------------------------------------------
// QKV GEMM: [4096,1024] x [1024,3072] -> scatter into q, k, v^T (bf16)
// Block = 4 waves; block tile 16 rows x 256 cols; wave tile 16x64.
// Ping-pong k-loop: A-register set computes while B-register set loads,
// then roles swap. No copy rotation, no in-loop conditionals.
// ---------------------------------------------------------------------------

__global__ __launch_bounds__(128) void k_gemm_qkv(
    const __bf16* __restrict__ A,      // [B*N, D]
    const __bf16* __restrict__ Bt,     // [3D, D] (W_qkv^T)
    const float*  __restrict__ bias,   // [3D]
    __bf16* __restrict__ qb,           // [B,H,N,HD]
    __bf16* __restrict__ kb,           // [B,H,N,HD]
    __bf16* __restrict__ vt)           // [B,H,HD,N]
{
    const int wave = threadIdx.x >> 5;
    const int lane = threadIdx.x & 31;
    const int half = lane >> 4;
    const int nlo  = lane & 15;

    const __bf16* arow = A + (size_t)(blockIdx.x * 16) * D_MODEL;
    const int cbase = blockIdx.y * 256 + wave * 64;

    f32x8 acc[4] = {zero8(), zero8(), zero8(), zero8()};

    bf16x16 afA, afB, bfA[4], bfB[4];
    afA = load_frag(arow, D_MODEL);
#pragma unroll
    for (int t = 0; t < 4; ++t)
        bfA[t] = load_frag(Bt + (size_t)(cbase + t * 16) * D_MODEL, D_MODEL);

    const int KI = D_MODEL / 32;                 // 32 k-chunks (even)
    for (int ki = 0; ki < KI - 2; ki += 2) {
        afB = load_frag(arow + (ki + 1) * 32, D_MODEL);
#pragma unroll
        for (int t = 0; t < 4; ++t)
            bfB[t] = load_frag(Bt + (size_t)(cbase + t * 16) * D_MODEL + (ki + 1) * 32, D_MODEL);
#pragma unroll
        for (int t = 0; t < 4; ++t)
            acc[t] = wmma_bf16(afA, bfA[t], acc[t]);

        afA = load_frag(arow + (ki + 2) * 32, D_MODEL);
#pragma unroll
        for (int t = 0; t < 4; ++t)
            bfA[t] = load_frag(Bt + (size_t)(cbase + t * 16) * D_MODEL + (ki + 2) * 32, D_MODEL);
#pragma unroll
        for (int t = 0; t < 4; ++t)
            acc[t] = wmma_bf16(afB, bfB[t], acc[t]);
    }
    {   // tail: chunks KI-2 (already in A-regs) and KI-1
        afB = load_frag(arow + (KI - 1) * 32, D_MODEL);
#pragma unroll
        for (int t = 0; t < 4; ++t)
            bfB[t] = load_frag(Bt + (size_t)(cbase + t * 16) * D_MODEL + (KI - 1) * 32, D_MODEL);
#pragma unroll
        for (int t = 0; t < 4; ++t)
            acc[t] = wmma_bf16(afA, bfA[t], acc[t]);
#pragma unroll
        for (int t = 0; t < 4; ++t)
            acc[t] = wmma_bf16(afB, bfB[t], acc[t]);
    }

#pragma unroll
    for (int t = 0; t < 4; ++t) {
#pragma unroll
        for (int r = 0; r < 8; ++r) {
            const int i = blockIdx.x * 16 + r + 8 * half;  // token row in [0, B*N)
            const int c = cbase + t * 16 + nlo;            // output channel in [0, 3D)
            const float v = acc[t][r] + bias[c];
            const int bb   = i >> 11;            // / N_SEQ
            const int nIdx = i & (N_SEQ - 1);
            const int mat  = c >> 10;            // 0=q 1=k 2=v
            const int rest = c & (D_MODEL - 1);
            const int hh   = rest >> 6;
            const int dd   = rest & 63;
            const size_t bh = (size_t)bb * H_NUM + hh;
            if (mat == 0)      qb[(bh * N_SEQ + nIdx) * HD + dd] = (__bf16)v;
            else if (mat == 1) kb[(bh * N_SEQ + nIdx) * HD + dd] = (__bf16)v;
            else               vt[(bh * HD + dd) * N_SEQ + nIdx] = (__bf16)v;
        }
    }
}

// ---------------------------------------------------------------------------
// Flash attention with distance bias. 1 wave per (b, h, 16-query tile).
// K/V tiles are async-copied into double-buffered LDS (ASYNCcnt pipeline);
// per 32-key step: 4 WMMAs for S, online softmax (exp2 domain), 4 WMMAs for O.
// ---------------------------------------------------------------------------

// Copy one 32-key tile of K ([32,HD] contiguous, 4 KB) and V^T ([HD rows, 32 keys],
// 64 B per row, row stride N_SEQ*2 B) into LDS. 16 b128 async ops total.
__device__ __forceinline__ void prefetch_kv(const __bf16* kglob, const __bf16* vglob,
                                            __bf16* kl, __bf16* vl, int lane) {
#pragma unroll
    for (int j = 0; j < 8; ++j) {   // K: 8 x (32 lanes x 16B) = 4 KB linear
        async_copy16((const char*)kglob + j * 512 + lane * 16,
                     (char*)kl + j * 512 + lane * 16);
    }
    const int r4  = lane >> 2;      // 0..7 : row within 8-row group
    const int c16 = (lane & 3) * 16;
#pragma unroll
    for (int j = 0; j < 8; ++j) {   // V^T: 8 rows per op x 8 = 64 rows
        const int row = j * 8 + r4;
        async_copy16((const char*)vglob + (size_t)row * (N_SEQ * 2) + c16,
                     (char*)vl + row * 64 + c16);
    }
}

__global__ __launch_bounds__(32) void k_flash(
    const __bf16* __restrict__ qb,     // [B,H,N,HD]
    const __bf16* __restrict__ kb,     // [B,H,N,HD]
    const __bf16* __restrict__ vt,     // [B,H,HD,N]
    const float*  __restrict__ gamma,  // [B,H]
    const float*  __restrict__ dist,   // [N,N]
    __bf16* __restrict__ ob,           // [B,N,H*HD]
    float scale)
{
    __shared__ __align__(16) __bf16 kl[2][32 * HD];   // 2 x 4 KB
    __shared__ __align__(16) __bf16 vl[2][HD * 32];   // 2 x 4 KB
    __shared__ __align__(16) __bf16 pbuf[16 * 32];    // P re-fragment staging

    const int lane = threadIdx.x & 31;
    const int half = lane >> 4;
    const int nlo  = lane & 15;
    const int qt = blockIdx.x;
    const int h  = blockIdx.y;
    const int b  = blockIdx.z;

    const size_t bh = (size_t)b * H_NUM + h;
    const __bf16* qbase = qb + (bh * N_SEQ + (size_t)qt * 16) * HD;
    const __bf16* kbase = kb + bh * N_SEQ * HD;
    const __bf16* vbase = vt + bh * HD * N_SEQ;
    const float* drow = dist + (size_t)(qt * 16) * N_SEQ;

    // exp2-domain softmax: softmax(x) == softmax2(x * log2(e)); each
    // exponential is then a single v_exp_f32.
    const float LOG2E = 1.4426950408889634f;
    const float sc2 = scale * LOG2E;
    const float g2  = gamma[b * H_NUM + h] * LOG2E;

    // Q fragments (k-chunks 0..31 and 32..63 of hd=64)
    const bf16x16 qa0 = load_frag(qbase, HD);
    const bf16x16 qa1 = load_frag(qbase + 32, HD);

    f32x8 o0 = zero8(), o1 = zero8(), o2 = zero8(), o3 = zero8();
    float mrow[8], lrow[8];
#pragma unroll
    for (int r = 0; r < 8; ++r) { mrow[r] = -INFINITY; lrow[r] = 0.f; }

    // prime the pipeline: tile 0 into buffer 0
    prefetch_kv(kbase, vbase, kl[0], vl[0], lane);

    for (int k0 = 0; k0 < N_SEQ; k0 += 32) {
        const int cur = (k0 >> 5) & 1;
        if (k0 + 32 < N_SEQ) {
            // issue next tile's copies, then wait for the current tile only
            prefetch_kv(kbase + (size_t)(k0 + 32) * HD, vbase + (k0 + 32),
                        kl[cur ^ 1], vl[cur ^ 1], lane);
            WAIT_ASYNC_PREV();
        } else {
            WAIT_ASYNC_ALL();
        }
        const __bf16* kcur = kl[cur];
        const __bf16* vcur = vl[cur];

        // ---- S = Q K^T (16x32 tile as two 16x16 accumulators) ----
        f32x8 s0 = zero8(), s1 = zero8();
        s0 = wmma_bf16(qa0, load_frag(kcur,            HD), s0);
        s0 = wmma_bf16(qa1, load_frag(kcur + 32,       HD), s0);
        s1 = wmma_bf16(qa0, load_frag(kcur + 16 * HD,      HD), s1);
        s1 = wmma_bf16(qa1, load_frag(kcur + 16 * HD + 32, HD), s1);

        // ---- bias + online softmax (exp2 domain) ----
        float p0[8], p1[8], alpha[8];
#pragma unroll
        for (int r = 0; r < 8; ++r) {
            const int m = r + 8 * half;                 // query row within tile
            const float d0 = drow[(size_t)m * N_SEQ + k0 + nlo];
            const float d1 = drow[(size_t)m * N_SEQ + k0 + 16 + nlo];
            const float v0 = s0[r] * sc2 - g2 * d0;
            const float v1 = s1[r] * sc2 - g2 * d1;
            const float mx   = hmax16(fmaxf(v0, v1));
            const float mnew = fmaxf(mrow[r], mx);
            const float a  = exp2f(mrow[r] - mnew);
            const float e0 = exp2f(v0 - mnew);
            const float e1 = exp2f(v1 - mnew);
            lrow[r] = lrow[r] * a + hsum16(e0 + e1);
            mrow[r] = mnew;
            alpha[r] = a; p0[r] = e0; p1[r] = e1;
        }

        // ---- re-fragment P through LDS (accumulator layout -> A layout) ----
        __syncthreads();
#pragma unroll
        for (int r = 0; r < 8; ++r) {
            const int m = r + 8 * half;
            pbuf[m * 32 + nlo]      = (__bf16)p0[r];
            pbuf[m * 32 + 16 + nlo] = (__bf16)p1[r];
        }
        __syncthreads();
        const bf16x16 pa = load_frag(pbuf, 32);

        // ---- O = O*alpha + P V ----
#pragma unroll
        for (int r = 0; r < 8; ++r) {
            const float a = alpha[r];
            o0[r] *= a; o1[r] *= a; o2[r] *= a; o3[r] *= a;
        }
        o0 = wmma_bf16(pa, load_frag(vcur + 0 * 16 * 32, 32), o0);
        o1 = wmma_bf16(pa, load_frag(vcur + 1 * 16 * 32, 32), o1);
        o2 = wmma_bf16(pa, load_frag(vcur + 2 * 16 * 32, 32), o2);
        o3 = wmma_bf16(pa, load_frag(vcur + 3 * 16 * 32, 32), o3);
    }

    // ---- epilogue: normalize and store [B,N,H*HD] bf16 ----
    __bf16* obase = ob + ((size_t)b * N_SEQ + (size_t)qt * 16) * D_MODEL + (size_t)h * HD;
#pragma unroll
    for (int r = 0; r < 8; ++r) {
        const int m = r + 8 * half;
        const float inv = 1.0f / lrow[r];
        obase[(size_t)m * D_MODEL +  0 + nlo] = (__bf16)(o0[r] * inv);
        obase[(size_t)m * D_MODEL + 16 + nlo] = (__bf16)(o1[r] * inv);
        obase[(size_t)m * D_MODEL + 32 + nlo] = (__bf16)(o2[r] * inv);
        obase[(size_t)m * D_MODEL + 48 + nlo] = (__bf16)(o3[r] * inv);
    }
}

// ---------------------------------------------------------------------------
// Output projection: [4096,1024] x [1024,1024] -> f32 d_out (+ bias)
// Same ping-pong structure as the QKV GEMM.
// ---------------------------------------------------------------------------

__global__ __launch_bounds__(128) void k_gemm_proj(
    const __bf16* __restrict__ A,      // [B*N, D] (attention output, bf16)
    const __bf16* __restrict__ Bt,     // [D, D] (W_proj^T)
    const float*  __restrict__ bias,   // [D]
    float* __restrict__ out)           // [B*N, D] f32
{
    const int wave = threadIdx.x >> 5;
    const int lane = threadIdx.x & 31;
    const int half = lane >> 4;
    const int nlo  = lane & 15;

    const __bf16* arow = A + (size_t)(blockIdx.x * 16) * D_MODEL;
    const int cbase = blockIdx.y * 256 + wave * 64;

    f32x8 acc[4] = {zero8(), zero8(), zero8(), zero8()};

    bf16x16 afA, afB, bfA[4], bfB[4];
    afA = load_frag(arow, D_MODEL);
#pragma unroll
    for (int t = 0; t < 4; ++t)
        bfA[t] = load_frag(Bt + (size_t)(cbase + t * 16) * D_MODEL, D_MODEL);

    const int KI = D_MODEL / 32;
    for (int ki = 0; ki < KI - 2; ki += 2) {
        afB = load_frag(arow + (ki + 1) * 32, D_MODEL);
#pragma unroll
        for (int t = 0; t < 4; ++t)
            bfB[t] = load_frag(Bt + (size_t)(cbase + t * 16) * D_MODEL + (ki + 1) * 32, D_MODEL);
#pragma unroll
        for (int t = 0; t < 4; ++t)
            acc[t] = wmma_bf16(afA, bfA[t], acc[t]);

        afA = load_frag(arow + (ki + 2) * 32, D_MODEL);
#pragma unroll
        for (int t = 0; t < 4; ++t)
            bfA[t] = load_frag(Bt + (size_t)(cbase + t * 16) * D_MODEL + (ki + 2) * 32, D_MODEL);
#pragma unroll
        for (int t = 0; t < 4; ++t)
            acc[t] = wmma_bf16(afB, bfB[t], acc[t]);
    }
    {
        afB = load_frag(arow + (KI - 1) * 32, D_MODEL);
#pragma unroll
        for (int t = 0; t < 4; ++t)
            bfB[t] = load_frag(Bt + (size_t)(cbase + t * 16) * D_MODEL + (KI - 1) * 32, D_MODEL);
#pragma unroll
        for (int t = 0; t < 4; ++t)
            acc[t] = wmma_bf16(afA, bfA[t], acc[t]);
#pragma unroll
        for (int t = 0; t < 4; ++t)
            acc[t] = wmma_bf16(afB, bfB[t], acc[t]);
    }

#pragma unroll
    for (int t = 0; t < 4; ++t) {
#pragma unroll
        for (int r = 0; r < 8; ++r) {
            const int i = blockIdx.x * 16 + r + 8 * half;
            const int c = cbase + t * 16 + nlo;
            out[(size_t)i * D_MODEL + c] = acc[t][r] + bias[c];
        }
    }
}

// ---------------------------------------------------------------------------
// Launch
// ---------------------------------------------------------------------------

extern "C" void kernel_launch(void* const* d_in, const int* in_sizes, int n_in,
                              void* d_out, int out_size, void* d_ws, size_t ws_size,
                              hipStream_t stream) {
    (void)in_sizes; (void)n_in; (void)out_size; (void)ws_size;

    const float* x     = (const float*)d_in[0];   // [B,N,D]
    const float* gamma = (const float*)d_in[1];   // [B,H]
    const float* dist  = (const float*)d_in[2];   // [N,N]
    const float* Wqkv  = (const float*)d_in[3];   // [D,3D]
    const float* bqkv  = (const float*)d_in[4];   // [3D]
    const float* Wproj = (const float*)d_in[5];   // [D,D]
    const float* bproj = (const float*)d_in[6];   // [D]
    float* out = (float*)d_out;

    char* ws = (char*)d_ws;
    // Workspace layout (48 MB total)
    __bf16* xbf    = (__bf16*)(ws + 0);           //  8 MB  [B*N, D]
    __bf16* wqkvT  = (__bf16*)(ws + (8u  << 20)); //  6 MB  [3D, D]
    __bf16* wprojT = (__bf16*)(ws + (14u << 20)); //  2 MB  [D, D]
    __bf16* qb     = (__bf16*)(ws + (16u << 20)); //  8 MB  [B,H,N,HD]
    __bf16* kb     = (__bf16*)(ws + (24u << 20)); //  8 MB  [B,H,N,HD]
    __bf16* vt     = (__bf16*)(ws + (32u << 20)); //  8 MB  [B,H,HD,N]
    __bf16* ob     = (__bf16*)(ws + (40u << 20)); //  8 MB  [B,N,D]

    const int nx = B_DIM * N_SEQ * D_MODEL;
    k_cvt_bf16<<<(nx + 255) / 256, 256, 0, stream>>>(x, xbf, nx);

    const int nwq = D_MODEL * 3 * D_MODEL;
    k_transpose_cvt<<<(nwq + 255) / 256, 256, 0, stream>>>(Wqkv, wqkvT, D_MODEL, 3 * D_MODEL);
    const int nwp = D_MODEL * D_MODEL;
    k_transpose_cvt<<<(nwp + 255) / 256, 256, 0, stream>>>(Wproj, wprojT, D_MODEL, D_MODEL);

    dim3 gq(B_DIM * N_SEQ / 16, 3 * D_MODEL / 256);
    k_gemm_qkv<<<gq, 128, 0, stream>>>(xbf, wqkvT, bqkv, qb, kb, vt);

    const float scale = 1.0f / sqrtf((float)D_MODEL);
    dim3 gf(N_SEQ / 16, H_NUM, B_DIM);
    k_flash<<<gf, 32, 0, stream>>>(qb, kb, vt, gamma, dist, ob, scale);

    dim3 gp(B_DIM * N_SEQ / 16, D_MODEL / 256);
    k_gemm_proj<<<gp, 128, 0, stream>>>(ob, wprojT, bproj, out);
}
